// APPNP_5858335392241
// MI455X (gfx1250) — matrix operand order repaired and checked by reading, not compiled, not run on hardware
//
#include <hip/hip_runtime.h>

#define IN_C 512
#define HID  128
#define OUTC 16
#define ALPHA 0.1f

typedef __attribute__((ext_vector_type(16))) _Float16 v16h;
typedef __attribute__((ext_vector_type(8)))  _Float16 v8h;
typedef __attribute__((ext_vector_type(8)))  float    v8f;

// ---------------- degree / symmetric norm ----------------
__global__ void appnp_deg_init(float* deg, int n) {
    int i = blockIdx.x * blockDim.x + threadIdx.x;
    if (i < n) deg[i] = 1.0f;   // self-loop contributes 1
}

__global__ void appnp_deg_accum(const int* __restrict__ eidx, int E, float* deg) {
    int e = blockIdx.x * blockDim.x + threadIdx.x;
    if (e < E) {
        int d = eidx[E + e];    // edge_index[1][e]
        unsafeAtomicAdd(&deg[d], 1.0f);
    }
}

__global__ void appnp_dinv(float* deg, int n) {
    int i = blockIdx.x * blockDim.x + threadIdx.x;
    if (i < n) {
        float d = deg[i];
        deg[i] = (d > 0.0f) ? rsqrtf(d) : 0.0f;   // in-place: deg -> dinv
    }
}

// ---------------- fused MLP via WMMA ----------------
// h0 = relu(x @ W1 + b1) @ W2 + b2 ; also h = h0
__global__ void __launch_bounds__(256, 1)
appnp_mlp_wmma(const float* __restrict__ x,
               const float* __restrict__ W1, const float* __restrict__ b1,
               const float* __restrict__ W2, const float* __restrict__ b2,
               float* __restrict__ h0, float* __restrict__ h,
               int N, int nTiles) {
    // W1 transposed to [col][k] in f16 so B-operand reads are contiguous 32B
    __shared__ __align__(32) _Float16 lds_w1[HID * IN_C];   // 128 KB
    __shared__ __align__(32) _Float16 lds_w2[OUTC * HID];   // 4 KB
    __shared__ float lds_b1[HID];
    __shared__ float lds_b2[OUTC];
    __shared__ __align__(32) _Float16 lds_h[8 * 16 * HID];  // per-wave 16x128 stage, 32 KB

    for (int idx = threadIdx.x; idx < HID * IN_C; idx += blockDim.x) {
        int k = idx >> 7, c = idx & (HID - 1);              // W1 is [k][c] row-major
        lds_w1[c * IN_C + k] = (_Float16)W1[idx];
    }
    for (int idx = threadIdx.x; idx < OUTC * HID; idx += blockDim.x) {
        int k = idx >> 4, c = idx & (OUTC - 1);             // W2 is [k][c] row-major
        lds_w2[c * HID + k] = (_Float16)W2[idx];
    }
    if (threadIdx.x < HID)  lds_b1[threadIdx.x] = b1[threadIdx.x];
    if (threadIdx.x < OUTC) lds_b2[threadIdx.x] = b2[threadIdx.x];
    __syncthreads();

    const int wave = threadIdx.x >> 5;
    const int lane = threadIdx.x & 31;
    const int ln15 = lane & 15;
    const bool hiHalf = lane >= 16;
    const int kbA = hiHalf ? 16 : 0;    // A layout: lo lanes K 0..15; hi lanes K 16..31
    const int kbB = hiHalf ? 16 : 0;    // B layout: lo lanes K 0..15; hi lanes K 16..31
    _Float16* hs = &lds_h[wave * 16 * HID];

    for (int t = blockIdx.x * 8 + wave; t < nTiles; t += gridDim.x * 8) {
        const int rowbase = t * 16;
        int arow = rowbase + ln15;
        if (arow > N - 1) arow = N - 1;                     // keep EXEC full for WMMA
        const float* xr = x + (size_t)arow * IN_C;
        __builtin_prefetch(xr, 0, 1);

        v8f acc[8] = {};
        #pragma unroll 1
        for (int k0 = 0; k0 < IN_C; k0 += 32) {
            // A operand: 16x32 f16 tile from x (converted on the fly)
            const float4 f0 = *(const float4*)(xr + k0 + kbA + 0);
            const float4 f1 = *(const float4*)(xr + k0 + kbA + 4);
            const float4 f2 = *(const float4*)(xr + k0 + kbA + 8);
            const float4 f3 = *(const float4*)(xr + k0 + kbA + 12);
            v16h a;
            a[0]  = (_Float16)f0.x; a[1]  = (_Float16)f0.y; a[2]  = (_Float16)f0.z; a[3]  = (_Float16)f0.w;
            a[4]  = (_Float16)f1.x; a[5]  = (_Float16)f1.y; a[6]  = (_Float16)f1.z; a[7]  = (_Float16)f1.w;
            a[8]  = (_Float16)f2.x; a[9]  = (_Float16)f2.y; a[10] = (_Float16)f2.z; a[11] = (_Float16)f2.w;
            a[12] = (_Float16)f3.x; a[13] = (_Float16)f3.y; a[14] = (_Float16)f3.z; a[15] = (_Float16)f3.w;
            #pragma unroll
            for (int ct = 0; ct < 8; ct++) {
                const v16h bm = *(const v16h*)&lds_w1[(ct * 16 + ln15) * IN_C + k0 + kbB];
                acc[ct] = __builtin_amdgcn_wmma_f32_16x16x32_f16(
                    false, a, false, bm, (short)0, acc[ct], false, false);
            }
        }

        // bias + ReLU, restage 16x128 tile into LDS in [m][col] f16
        #pragma unroll
        for (int ct = 0; ct < 8; ct++) {
            const float bb = lds_b1[ct * 16 + ln15];
            #pragma unroll
            for (int j = 0; j < 8; j++) {
                const int m = hiHalf ? (j + 8) : j;
                float v = acc[ct][j] + bb;
                v = v > 0.0f ? v : 0.0f;
                hs[m * HID + ct * 16 + ln15] = (_Float16)v;
            }
        }

        // GEMM2: (16x128) @ (128x16), K in 4 WMMA steps
        v8f acc2 = {};
        #pragma unroll
        for (int ks = 0; ks < 4; ks++) {
            const int kk = ks * 32;
            const v8h alo = *(const v8h*)&hs[ln15 * HID + kk + kbA];
            const v8h ahi = *(const v8h*)&hs[ln15 * HID + kk + kbA + 8];
            v16h a2;
            #pragma unroll
            for (int i = 0; i < 8; i++) { a2[i] = alo[i]; a2[i + 8] = ahi[i]; }
            const v16h bm2 = *(const v16h*)&lds_w2[ln15 * HID + kk + kbB];
            acc2 = __builtin_amdgcn_wmma_f32_16x16x32_f16(
                false, a2, false, bm2, (short)0, acc2, false, false);
        }

        const float bb2 = lds_b2[ln15];
        #pragma unroll
        for (int j = 0; j < 8; j++) {
            const int row = rowbase + (hiHalf ? (j + 8) : j);
            if (row < N) {
                const float v = acc2[j] + bb2;
                h0[(size_t)row * OUTC + ln15] = v;
                h [(size_t)row * OUTC + ln15] = v;
            }
        }
    }
}

// ---------------- APPNP propagation ----------------
__global__ void appnp_zero(float* __restrict__ p, long long n) {
    long long i = (long long)blockIdx.x * blockDim.x + threadIdx.x;
    if (i < n) p[i] = 0.0f;
}

// thread = (edge, channel); 16 consecutive threads share an edge -> coalesced rows
__global__ void appnp_scatter(const int* __restrict__ eidx, const float* __restrict__ dinv,
                              const float* __restrict__ h, float* __restrict__ agg, int E) {
    long long idx = (long long)blockIdx.x * blockDim.x + threadIdx.x;
    if (idx < (long long)E * OUTC) {
        const int e = (int)(idx >> 4);
        const int c = (int)(idx & (OUTC - 1));
        const int s = eidx[e];
        const int d = eidx[E + e];
        const float w = dinv[s] * dinv[d];
        unsafeAtomicAdd(&agg[(size_t)d * OUTC + c], w * h[(size_t)s * OUTC + c]);
    }
}

// h_out = (1-a)*(agg + dinv[i]^2 * h_in) + a*h0   (self-loop folded in)
__global__ void appnp_combine(const float* __restrict__ agg, const float* __restrict__ h_in,
                              const float* __restrict__ h0, const float* __restrict__ dinv,
                              float* __restrict__ h_out, long long n) {
    long long idx = (long long)blockIdx.x * blockDim.x + threadIdx.x;
    if (idx < n) {
        const int i = (int)(idx >> 4);
        const float di = dinv[i];
        const float self = di * di * h_in[idx];
        h_out[idx] = (1.0f - ALPHA) * (agg[idx] + self) + ALPHA * h0[idx];
    }
}

extern "C" void kernel_launch(void* const* d_in, const int* in_sizes, int n_in,
                              void* d_out, int out_size, void* d_ws, size_t ws_size,
                              hipStream_t stream) {
    const float* x  = (const float*)d_in[0];
    const int*   ei = (const int*)d_in[1];
    const float* W1 = (const float*)d_in[2];
    const float* b1 = (const float*)d_in[3];
    const float* W2 = (const float*)d_in[4];
    const float* b2 = (const float*)d_in[5];
    float* out = (float*)d_out;

    const int N = in_sizes[0] / IN_C;
    const int E = in_sizes[1] / 2;

    float* ws   = (float*)d_ws;
    float* dinv = ws;                          // N floats (deg -> dinv in place)
    float* h0   = dinv + N;                    // N*16
    float* h    = h0 + (size_t)N * OUTC;       // N*16
    float* agg  = h  + (size_t)N * OUTC;       // N*16

    // degree + normalization
    appnp_deg_init <<<(N + 255) / 256, 256, 0, stream>>>(dinv, N);
    appnp_deg_accum<<<(E + 255) / 256, 256, 0, stream>>>(ei, E, dinv);
    appnp_dinv     <<<(N + 255) / 256, 256, 0, stream>>>(dinv, N);

    // fused MLP (WMMA)
    const int nTiles = (N + 15) / 16;
    int mlpBlocks = (nTiles + 7) / 8;
    if (mlpBlocks > 256) mlpBlocks = 256;
    appnp_mlp_wmma<<<mlpBlocks, 256, 0, stream>>>(x, W1, b1, W2, b2, h0, h, N, nTiles);

    // APPNP iterations
    const long long totN = (long long)N * OUTC;
    const long long totE = (long long)E * OUTC;
    const int zb = (int)((totN + 255) / 256);
    const int sb = (int)((totE + 255) / 256);
    for (int it = 0; it < 10; it++) {
        appnp_zero   <<<zb, 256, 0, stream>>>(agg, totN);
        appnp_scatter<<<sb, 256, 0, stream>>>(ei, dinv, h, agg, E);
        appnp_combine<<<zb, 256, 0, stream>>>(agg, h, h0, dinv, (it == 9) ? out : h, totN);
    }
}
